// RGCNLayer_25074019074324
// MI455X (gfx1250) — compile-verified, matrix operand order checked by reading, not run on hardware
//
#include <hip/hip_runtime.h>
#include <hip/hip_bf16.h>

// ---- problem constants (from reference) ----
#define N_NODES  10000
#define N_EDGES  320000
#define NUM_RELS 20
#define IN_FEAT  256
#define OUT_FEAT 256

typedef __attribute__((ext_vector_type(16))) __bf16 v16bf;
typedef __attribute__((ext_vector_type(8)))  __bf16 v8bf;
typedef __attribute__((ext_vector_type(4)))  __bf16 v4bf;
typedef __attribute__((ext_vector_type(8)))  float  v8f;

// ---------------------------------------------------------------------------
// 1) f32 -> bf16 convert of node features
// ---------------------------------------------------------------------------
__global__ void __launch_bounds__(256)
cvt_h_bf16(const float* __restrict__ h, __bf16* __restrict__ hb, int n)
{
    int i = blockIdx.x * 256 + threadIdx.x;
    if (i < n) hb[i] = (__bf16)h[i];
}

// ---------------------------------------------------------------------------
// 2) f32 -> bf16 convert + transpose of weights: Wt[r][out][in] = W[r][in][out]
//    so that each WMMA B-fragment lane (fixed out-column) loads K-contiguous.
// ---------------------------------------------------------------------------
__global__ void __launch_bounds__(256)
cvt_wt_bf16(const float* __restrict__ w, __bf16* __restrict__ wt)
{
    int i = blockIdx.x * 256 + threadIdx.x;           // over R*IN*OUT
    if (i >= NUM_RELS * IN_FEAT * OUT_FEAT) return;
    int r   = i / (IN_FEAT * OUT_FEAT);
    int rem = i % (IN_FEAT * OUT_FEAT);
    int k = rem / OUT_FEAT;                            // in index (coalesced read over n)
    int n = rem % OUT_FEAT;                            // out index
    wt[((size_t)r * OUT_FEAT + n) * IN_FEAT + k] = (__bf16)w[i];
}

// ---------------------------------------------------------------------------
// 3) zero the (poisoned) output accumulator
// ---------------------------------------------------------------------------
__global__ void __launch_bounds__(256)
zero_f32(float* __restrict__ p, int n)
{
    int i = blockIdx.x * 256 + threadIdx.x;
    if (i < n) p[i] = 0.0f;
}

// ---------------------------------------------------------------------------
// 4) WMMA grouped GEMM: h_all[m, r, n] = sum_k hb[m,k] * W[r][k][n]
//    bf16 inputs, f32 accumulate, bf16 output (102.4 MB h_all stays L2-resident).
//
//    Block = 8 waves sharing one (relation, 64-wide N-strip); the 64x256 bf16
//    weight strip (32 KB) is staged in LDS once per block, then each wave
//    computes a 16(M) x 64(N) strip (A from global, B from conflict-free LDS).
// ---------------------------------------------------------------------------
#define NT         4                      // 16-wide N-tiles per wave
#define MT_PER_BLK 8                      // M-tiles (waves) per block
#define M_TILES    (N_NODES / 16)         // 625
#define MG_BLKS    ((M_TILES + MT_PER_BLK - 1) / MT_PER_BLK)   // 79
#define LDS_PAD    8                      // bf16 elems: row stride 132 dwords == 4 mod 64

__global__ void __launch_bounds__(256)
rgcn_gemm_hall(const __bf16* __restrict__ hb,     // [N_NODES][IN_FEAT]
               const __bf16* __restrict__ wt,     // [R][OUT_FEAT][IN_FEAT]
               __bf16* __restrict__ h_all)        // [N_NODES][R][OUT_FEAT]
{
    __shared__ __bf16 lbs[NT * 16][IN_FEAT + LDS_PAD];   // 64 x 264 bf16 = 33 KB

    // block -> (mg, ng, rel)
    const int mg   = blockIdx.x % MG_BLKS;
    const int rest = blockIdx.x / MG_BLKS;
    const int ng   = rest % 4;                       // which 64-wide N strip
    const int rel  = rest / 4;                       // relation
    const int n0   = ng * (NT * 16);

    // --- cooperative stage of the B strip: Wt[rel][n0..n0+63][0..255] -> LDS ---
    // 64 rows x 256 bf16 = 2048 x 16B chunks; 256 threads x 8 iterations.
    #pragma unroll
    for (int it = 0; it < 8; ++it) {
        const int cid = it * 256 + threadIdx.x;
        const int row = cid >> 5;                    // 32 chunks per row
        const int off = (cid & 31) * 8;
        v8bf v = *(const v8bf*)(wt + ((size_t)rel * OUT_FEAT + n0 + row) * IN_FEAT + off);
        *(v8bf*)(&lbs[row][off]) = v;                // row stride 528B (16B aligned)
    }
    __syncthreads();

    const int lane = threadIdx.x & 31;
    const int half = lane >> 4;                      // K-half select (ISA A/B layout)
    const int l15  = lane & 15;

    const int mt = mg * MT_PER_BLK + (threadIdx.x >> 5);
    if (mt >= M_TILES) return;                       // after the barrier: safe
    const int m0 = mt * 16;

    v8f acc[NT] = {};                                // 4 x 16x16 f32 accumulators

    // A: lane = row M, VGPR0-3 hold K=base..base+7 (half-dependent), VGPR4-7 +16.
    const __bf16* arow = hb + (size_t)(m0 + l15) * IN_FEAT + half * 8;

    for (int kk = 0; kk < IN_FEAT; kk += 32) {
        v8bf alo = *(const v8bf*)(arow + kk);        // 16B global load
        v8bf ahi = *(const v8bf*)(arow + kk + 16);
        v16bf a;
        #pragma unroll
        for (int i = 0; i < 8; ++i) { a[i] = alo[i]; a[8 + i] = ahi[i]; }

        #pragma unroll
        for (int t = 0; t < NT; ++t) {
            // B fragment from LDS: lane = column N, same K packing as A.
            // Bank-conflict-free: row stride 132 dwords => lane c hits banks 4c..4c+3.
            const __bf16* bcol = &lbs[t * 16 + l15][kk + half * 8];
            v8bf blo = *(const v8bf*)(bcol);
            v8bf bhi = *(const v8bf*)(bcol + 16);
            v16bf b;
            #pragma unroll
            for (int i = 0; i < 8; ++i) { b[i] = blo[i]; b[8 + i] = bhi[i]; }

            acc[t] = __builtin_amdgcn_wmma_f32_16x16x32_bf16(
                         /*neg_a=*/false, a, /*neg_b=*/false, b,
                         /*c_mod=*/(short)0, acc[t],
                         /*reuse_a=*/false, /*reuse_b=*/false);
        }
    }

    // C/D layout: lane L, VGPR v -> M = v + 8*(L/16), N = L&15.
    #pragma unroll
    for (int t = 0; t < NT; ++t) {
        const int n = n0 + t * 16 + l15;
        #pragma unroll
        for (int v = 0; v < 8; ++v) {
            const int m = m0 + v + 8 * half;
            h_all[((size_t)m * NUM_RELS + rel) * OUT_FEAT + n] = (__bf16)acc[t][v];
        }
    }
}

// ---------------------------------------------------------------------------
// 5) edge gather + scale + atomic scatter-add (f32 accumulation in d_out).
//    64 lanes per edge, 4 channels per lane (8B bf16 load, 4 f32 atomics).
//    h_all (102 MB bf16) is L2-resident, so gathers hit the 192 MB L2.
// ---------------------------------------------------------------------------
__global__ void __launch_bounds__(256)
rgcn_edge_scatter(const __bf16* __restrict__ h_all,
                  const float*  __restrict__ norm,
                  const int*    __restrict__ esrc,
                  const int*    __restrict__ edst,
                  const int*    __restrict__ erel,
                  float*        __restrict__ out)
{
    const int e = blockIdx.x * 4 + (threadIdx.x >> 6);
    if (e >= N_EDGES) return;
    const int c = (threadIdx.x & 63) * 4;

    const int   src = esrc[e];
    const int   dst = edst[e];
    const int   r   = erel[e];
    const float nw  = norm[e];

    const v4bf m = *(const v4bf*)(h_all + ((size_t)src * NUM_RELS + r) * OUT_FEAT + c);
    float* o = out + (size_t)dst * OUT_FEAT + c;
    #pragma unroll
    for (int i = 0; i < 4; ++i)
        atomicAdd(o + i, (float)m[i] * nw);          // global_atomic_add_f32, no return
}

// ---------------------------------------------------------------------------
// 6) ReLU
// ---------------------------------------------------------------------------
__global__ void __launch_bounds__(256)
relu_f32(float* __restrict__ p, int n)
{
    int i = blockIdx.x * 256 + threadIdx.x;
    if (i < n) p[i] = fmaxf(p[i], 0.0f);
}

// ---------------------------------------------------------------------------
// launch
// ---------------------------------------------------------------------------
extern "C" void kernel_launch(void* const* d_in, const int* in_sizes, int n_in,
                              void* d_out, int out_size, void* d_ws, size_t ws_size,
                              hipStream_t stream)
{
    const float* h    = (const float*)d_in[0];   // [N_NODES, IN_FEAT]
    const float* w    = (const float*)d_in[1];   // [R, IN_FEAT, OUT_FEAT]
    const float* norm = (const float*)d_in[2];   // [N_EDGES, 1]
    const int*   esrc = (const int*)  d_in[3];
    const int*   edst = (const int*)  d_in[4];
    const int*   erel = (const int*)  d_in[5];
    float*       out  = (float*)d_out;           // [N_NODES, OUT_FEAT]

    // workspace layout (bytes):
    //   h_all bf16 : N*R*OUT*2      = 102,400,000
    //   hb    bf16 : N*IN*2         =   5,120,000
    //   wt    bf16 : R*OUT*IN*2     =   2,621,440
    char* ws = (char*)d_ws;
    __bf16* hall = (__bf16*)ws;
    __bf16* hb   = (__bf16*)(ws + (size_t)N_NODES * NUM_RELS * OUT_FEAT * 2);
    __bf16* wt   = (__bf16*)(ws + (size_t)N_NODES * NUM_RELS * OUT_FEAT * 2
                                + (size_t)N_NODES * IN_FEAT * 2);

    const int nh = N_NODES * IN_FEAT;                 // 2,560,000
    const int nw = NUM_RELS * IN_FEAT * OUT_FEAT;     // 1,310,720
    const int no = N_NODES * OUT_FEAT;                // 2,560,000

    cvt_h_bf16 <<<(nh + 255) / 256, 256, 0, stream>>>(h, hb, nh);
    cvt_wt_bf16<<<(nw + 255) / 256, 256, 0, stream>>>(w, wt);
    zero_f32   <<<(no + 255) / 256, 256, 0, stream>>>(out, no);

    // blocks = ceil(625/8) M-groups * 4 N-strips * 20 relations = 6320
    const int gemm_blocks = MG_BLKS * 4 * NUM_RELS;
    rgcn_gemm_hall<<<gemm_blocks, 256, 0, stream>>>(hb, wt, hall);

    rgcn_edge_scatter<<<(N_EDGES + 3) / 4, 256, 0, stream>>>(hall, norm, esrc, edst,
                                                             erel, out);

    relu_f32<<<(no + 255) / 256, 256, 0, stream>>>(out, no);
}
